// ShapesSender_60060822667888
// MI455X (gfx1250) — matrix operand order repaired and checked by reading, not compiled, not run on hardware
//
#include <hip/hip_runtime.h>
#include <hip/hip_bf16.h>
#include <math.h>

// Problem constants
#define B_   2048
#define V_   4096
#define L_   10
#define H_   512
#define E_   256
#define META_ 15
#define G4_  2048   // 4*H

typedef __attribute__((ext_vector_type(16))) __bf16 v16bf;
typedef __attribute__((ext_vector_type(8)))  float  v8f;

union Frag16 { v16bf v; uint4 u[2]; };

// CDNA5 async direct-to-LDS 16B load (ASYNCcnt-tracked).
// vdst = per-lane LDS byte address, vaddr = 64-bit global address.
__device__ __forceinline__ void async_ld16(const void* g, unsigned lds_off) {
  asm volatile("global_load_async_to_lds_b128 %0, %1, off"
               :: "v"(lds_off), "v"((unsigned long long)(uintptr_t)g)
               : "memory");
}

// ---------------------------------------------------------------------------
// BF16 WMMA GEMM:  C[M,N] = A[M,K] @ Bt[N,K]^T   (A row-major, Bt is B stored
// as [N][K] row-major, C f32 row-major). M,N multiples of 128, K mult of 32.
// Block = 256 threads = 8 waves; tile 128x128; wave computes 64x32.
// Double-buffered LDS fed by global_load_async_to_lds_b128.
// ---------------------------------------------------------------------------
__global__ __launch_bounds__(256)
void gemm_bf16_wmma(const __bf16* __restrict__ A, const __bf16* __restrict__ Bt,
                    float* __restrict__ C, int M, int N, int K) {
  // [buf][A=0 / B=1][128 rows * 20-dword padded pitch] = 40KB total
  __shared__ unsigned int lds[2][2][128 * 20];

  const int tid   = threadIdx.x;
  const int lane  = tid & 31;
  const int wave  = tid >> 5;
  const int waveM = wave >> 2;          // 0..1
  const int waveN = wave & 3;           // 0..3
  const int bm    = blockIdx.y * 128;
  const int bn    = blockIdx.x * 128;

  // staging decomposition: 512 16B-items per matrix per chunk; thread owns
  // items tid and tid+256 -> rows {tid>>2, (tid>>2)+64}, col-group tid&3
  const int srow = tid >> 2;            // 0..63
  const int scg  = tid & 3;             // 0..3

  v8f acc[4][2];
#pragma unroll
  for (int mt = 0; mt < 4; ++mt)
#pragma unroll
    for (int nt = 0; nt < 2; ++nt)
#pragma unroll
      for (int r = 0; r < 8; ++r) acc[mt][nt][r] = 0.0f;

  const int nk = K >> 5;                // #32-wide K chunks (>= 8 for our shapes)

  // prologue: stream chunks 0 and 1 into the two buffers (4 async ops each)
#pragma unroll
  for (int pc = 0; pc < 2; ++pc) {
    const int k0 = pc * 32;
#pragma unroll
    for (int i2 = 0; i2 < 2; ++i2) {
      const int r = srow + i2 * 64;
      async_ld16(A  + (size_t)(bm + r) * K + k0 + scg * 8,
                 (unsigned)(uintptr_t)&lds[pc][0][r * 20 + scg * 4]);
      async_ld16(Bt + (size_t)(bn + r) * K + k0 + scg * 8,
                 (unsigned)(uintptr_t)&lds[pc][1][r * 20 + scg * 4]);
    }
  }

  for (int i = 0; i < nk; ++i) {
    const unsigned int* ldsA = lds[i & 1][0];
    const unsigned int* ldsB = lds[i & 1][1];

    // chunk i complete when ASYNCcnt <= 4 (only chunk i+1 still in flight)
    if (i + 1 < nk) asm volatile("s_wait_asynccnt 0x4" ::: "memory");
    else            asm volatile("s_wait_asynccnt 0x0" ::: "memory");
    __syncthreads();                    // all waves' async data visible

    // A fragment: lane holds row M=(lane&15); K dword groups {0..3,8..11}
    // (lanes 0-15) or {4..7,12..15} (lanes 16-31)
    Frag16 afr[4];
    {
      const int m0   = waveM * 64 + (lane & 15);
      const int kb2a = (lane & 16) ? 4 : 0;
#pragma unroll
      for (int mt = 0; mt < 4; ++mt) {
        const int r = m0 + mt * 16;
        afr[mt].u[0] = *(const uint4*)&ldsA[r * 20 + kb2a];
        afr[mt].u[1] = *(const uint4*)&ldsA[r * 20 + kb2a + 8];
      }
    }
    // B fragment: lane holds column N=(lane&15); K dword groups {0..3,8..11}
    // (lanes 0-15) or {4..7,12..15} (lanes 16-31)
    Frag16 bfr[2];
    {
      const int n0   = waveN * 32 + (lane & 15);
      const int kb2b = (lane & 16) ? 4 : 0;
#pragma unroll
      for (int nt = 0; nt < 2; ++nt) {
        const int r = n0 + nt * 16;
        bfr[nt].u[0] = *(const uint4*)&ldsB[r * 20 + kb2b];
        bfr[nt].u[1] = *(const uint4*)&ldsB[r * 20 + kb2b + 8];
      }
    }
#pragma unroll
    for (int mt = 0; mt < 4; ++mt)
#pragma unroll
      for (int nt = 0; nt < 2; ++nt)
        acc[mt][nt] = __builtin_amdgcn_wmma_f32_16x16x32_bf16(
            false, afr[mt].v, false, bfr[nt].v, (short)0, acc[mt][nt], false, false);

    __syncthreads();                    // all waves done reading buf[i&1]

    // refill the just-freed buffer with chunk i+2
    if (i + 2 < nk) {
      const int k0 = (i + 2) * 32;
#pragma unroll
      for (int i2 = 0; i2 < 2; ++i2) {
        const int r = srow + i2 * 64;
        async_ld16(A  + (size_t)(bm + r) * K + k0 + scg * 8,
                   (unsigned)(uintptr_t)&lds[i & 1][0][r * 20 + scg * 4]);
        async_ld16(Bt + (size_t)(bn + r) * K + k0 + scg * 8,
                   (unsigned)(uintptr_t)&lds[i & 1][1][r * 20 + scg * 4]);
      }
    }
  }

  // store: C/D layout — VGPR r, lanes0-15: M=r, lanes16-31: M=r+8; N = lane&15
  const int nc    = lane & 15;
  const int rhalf = (lane >> 4) * 8;
#pragma unroll
  for (int mt = 0; mt < 4; ++mt)
#pragma unroll
    for (int nt = 0; nt < 2; ++nt) {
      const int gm0 = bm + waveM * 64 + mt * 16 + rhalf;
      const int gn  = bn + waveN * 32 + nt * 16 + nc;
#pragma unroll
      for (int r = 0; r < 8; ++r)
        C[(size_t)(gm0 + r) * N + gn] = acc[mt][nt][r];
    }
}

// ---------------------------------------------------------------------------
// Small helper kernels
// ---------------------------------------------------------------------------
__global__ void zero_vec4_kernel(float* __restrict__ p, size_t n4) {
  size_t i = (size_t)blockIdx.x * blockDim.x + threadIdx.x;
  size_t stride = (size_t)gridDim.x * blockDim.x;
  float4 z; z.x = 0.f; z.y = 0.f; z.z = 0.f; z.w = 0.f;
  for (; i < n4; i += stride) ((float4*)p)[i] = z;
}

__global__ void zero_scalar_kernel(float* __restrict__ p, size_t n) {
  size_t i = (size_t)blockIdx.x * blockDim.x + threadIdx.x;
  size_t stride = (size_t)gridDim.x * blockDim.x;
  for (; i < n; i += stride) p[i] = 0.f;
}

__global__ void f32_to_bf16_kernel(const float* __restrict__ in,
                                   __bf16* __restrict__ out, size_t n) {
  size_t i = (size_t)blockIdx.x * blockDim.x + threadIdx.x;
  size_t stride = (size_t)gridDim.x * blockDim.x;
  for (; i < n; i += stride) out[i] = (__bf16)in[i];
}

// W_out [K=512, N=4096] -> Wt bf16 [N][K]
__global__ void transpose_to_bf16_kernel(const float* __restrict__ in,
                                         __bf16* __restrict__ out, int K, int N) {
  size_t i = (size_t)blockIdx.x * blockDim.x + threadIdx.x;
  if (i >= (size_t)K * N) return;
  int k = (int)(i / N);
  int n = (int)(i % N);
  out[(size_t)n * K + k] = (__bf16)in[i];
}

// seq_len=11, idx=SOS(0), ent_b=0, output[b,0,SOS]=1
__global__ void init_state_kernel(float* __restrict__ out, int* __restrict__ seq_len,
                                  int* __restrict__ idx, float* __restrict__ ent_b) {
  int b = blockIdx.x * blockDim.x + threadIdx.x;
  if (b >= B_) return;
  seq_len[b] = L_ + 1;
  idx[b]     = 0;
  ent_b[b]   = 0.f;
  out[(size_t)b * (L_ + 1) * V_ + 0] = 1.0f;   // one-hot SOS at [b,0,0]
}

// h0 = hidden_state @ W_in + b_in ; c0 = 0
__global__ void h0_kernel(const float* __restrict__ hs, const float* __restrict__ W_in,
                          const float* __restrict__ b_in, float* __restrict__ h,
                          __bf16* __restrict__ hbf, float* __restrict__ c) {
  int i = blockIdx.x * blockDim.x + threadIdx.x;   // B*H
  if (i >= B_ * H_) return;
  int b = i >> 9, j = i & (H_ - 1);
  float acc = b_in[j];
#pragma unroll
  for (int m = 0; m < META_; ++m) acc += hs[b * META_ + m] * W_in[m * H_ + j];
  h[i] = acc; hbf[i] = (__bf16)acc; c[i] = 0.f;
}

// LSTM cell pointwise: gates = gates_h + Wx[idx] + b_ih + b_hh
__global__ void lstm_cell_kernel(const float* __restrict__ gates_h,
                                 const float* __restrict__ Wx,
                                 const int* __restrict__ idx,
                                 const float* __restrict__ b_ih,
                                 const float* __restrict__ b_hh,
                                 float* __restrict__ c, float* __restrict__ h,
                                 __bf16* __restrict__ hbf) {
  int i = blockIdx.x * blockDim.x + threadIdx.x;   // B*H
  if (i >= B_ * H_) return;
  int b = i >> 9, j = i & (H_ - 1);
  const float* g  = gates_h + (size_t)b * G4_;
  const float* wx = Wx + (size_t)idx[b] * G4_;
  float ig = g[j]            + wx[j]            + b_ih[j]            + b_hh[j];
  float fg = g[j + H_]       + wx[j + H_]       + b_ih[j + H_]       + b_hh[j + H_];
  float gg = g[j + 2 * H_]   + wx[j + 2 * H_]   + b_ih[j + 2 * H_]   + b_hh[j + 2 * H_];
  float og = g[j + 3 * H_]   + wx[j + 3 * H_]   + b_ih[j + 3 * H_]   + b_hh[j + 3 * H_];
  float si = 1.f / (1.f + __expf(-ig));
  float sf = 1.f / (1.f + __expf(-fg));
  float so = 1.f / (1.f + __expf(-og));
  float cn = sf * c[i] + si * tanhf(gg);
  float hn = so * tanhf(cn);
  c[i] = cn; h[i] = hn; hbf[i] = (__bf16)hn;
}

// embeds_out[b,t,:] = embedding[idx[b],:]
__global__ void embeds_kernel(const float* __restrict__ embedding,
                              const int* __restrict__ idx,
                              float* __restrict__ embeds_out, int t) {
  int i = blockIdx.x * blockDim.x + threadIdx.x;   // B*E
  if (i >= B_ * E_) return;
  int b = i >> 8, e = i & (E_ - 1);
  embeds_out[(size_t)b * L_ * E_ + (size_t)t * E_ + e] =
      embedding[(size_t)idx[b] * E_ + e];
}

// Row softmax stats + argmax(logits + gumbel); writes one-hot, seq_len, idx, entropy
__global__ __launch_bounds__(256)
void softmax_argmax_kernel(const float* __restrict__ logits,
                           const float* __restrict__ b_out,
                           const float* __restrict__ gumbel_t,  // [B,V] slice for step t
                           float* __restrict__ out,             // full output [B,L+1,V]
                           int* __restrict__ seq_len,
                           int* __restrict__ idx,
                           float* __restrict__ ent_b, int t) {
  __shared__ float redf[256];
  __shared__ int   redi[256];
  const int b   = blockIdx.x;
  const int tid = threadIdx.x;
  const float* lrow = logits   + (size_t)b * V_;
  const float* grow = gumbel_t + (size_t)b * V_;

  float lc[V_ / 256];
  float lmax = -INFINITY, gmax = -INFINITY;
  int gidx = 0x7fffffff;
#pragma unroll
  for (int i = 0; i < V_ / 256; ++i) {
    int v = tid + i * 256;
    float l = lrow[v] + b_out[v];
    lc[i] = l;
    lmax = fmaxf(lmax, l);
    float gy = l + grow[v];
    if (gy > gmax || (gy == gmax && v < gidx)) { gmax = gy; gidx = v; }
  }

  // max reduce
  redf[tid] = lmax; __syncthreads();
  for (int s = 128; s > 0; s >>= 1) {
    if (tid < s) redf[tid] = fmaxf(redf[tid], redf[tid + s]);
    __syncthreads();
  }
  const float m = redf[0];
  __syncthreads();

  // argmax reduce (first-index tie-break)
  redf[tid] = gmax; redi[tid] = gidx; __syncthreads();
  for (int s = 128; s > 0; s >>= 1) {
    if (tid < s) {
      float ov = redf[tid + s]; int oi = redi[tid + s];
      if (ov > redf[tid] || (ov == redf[tid] && oi < redi[tid])) {
        redf[tid] = ov; redi[tid] = oi;
      }
    }
    __syncthreads();
  }
  const int amax = redi[0];
  __syncthreads();

  // sum exp and sum l*exp
  float s1 = 0.f, s2 = 0.f;
#pragma unroll
  for (int i = 0; i < V_ / 256; ++i) {
    float e = __expf(lc[i] - m);
    s1 += e; s2 += e * lc[i];
  }
  redf[tid] = s1; __syncthreads();
  for (int s = 128; s > 0; s >>= 1) {
    if (tid < s) redf[tid] += redf[tid + s];
    __syncthreads();
  }
  const float S1 = redf[0];
  __syncthreads();
  redf[tid] = s2; __syncthreads();
  for (int s = 128; s > 0; s >>= 1) {
    if (tid < s) redf[tid] += redf[tid + s];
    __syncthreads();
  }
  const float S2 = redf[0];

  if (tid == 0) {
    float ent = (m + logf(S1)) - S2 / S1;       // lse - sum(p*logits)
    ent_b[b] += ent;                             // single writer per b: deterministic
    idx[b] = amax;
    out[(size_t)b * (L_ + 1) * V_ + (size_t)(t + 1) * V_ + amax] = 1.0f;
    if (amax == 0 && seq_len[b] == L_ + 1) seq_len[b] = t + 2;  // EOS==0
  }
}

__global__ __launch_bounds__(256)
void finalize_entropy_kernel(const float* __restrict__ ent_b, float* __restrict__ ent_out) {
  __shared__ float red[256];
  float s = 0.f;
  for (int i = threadIdx.x; i < B_; i += 256) s += ent_b[i];
  red[threadIdx.x] = s; __syncthreads();
  for (int st = 128; st > 0; st >>= 1) {
    if (threadIdx.x < st) red[threadIdx.x] += red[threadIdx.x + st];
    __syncthreads();
  }
  if (threadIdx.x == 0) ent_out[0] = red[0] / (float)(B_ * L_);
}

// ---------------------------------------------------------------------------
extern "C" void kernel_launch(void* const* d_in, const int* in_sizes, int n_in,
                              void* d_out, int out_size, void* d_ws, size_t ws_size,
                              hipStream_t stream) {
  (void)in_sizes; (void)n_in; (void)out_size; (void)ws_size;
  const float* hidden_state = (const float*)d_in[0];
  const float* W_in      = (const float*)d_in[1];
  const float* b_in      = (const float*)d_in[2];
  const float* embedding = (const float*)d_in[3];
  const float* W_ih      = (const float*)d_in[4];   // [4H,E] -> already [N][K]
  const float* W_hh      = (const float*)d_in[5];   // [4H,H] -> already [N][K]
  const float* b_ih      = (const float*)d_in[6];
  const float* b_hh      = (const float*)d_in[7];
  const float* W_out     = (const float*)d_in[8];   // [H,V]  -> transpose to [N][K]
  const float* b_out     = (const float*)d_in[9];
  const float* gumbel    = (const float*)d_in[10];  // [L,B,V]

  // Output layout (flat, return order)
  float* out          = (float*)d_out;
  const size_t N_OUT  = (size_t)B_ * (L_ + 1) * V_;           // 92274688
  int*   seq_len      = (int*)(out + N_OUT);                  // 2048 (int32 bits)
  float* ent_out      = out + N_OUT + B_;                     // 1
  float* embeds_out   = out + N_OUT + B_ + 1;                 // B*L*E
  float* sent_prob    = embeds_out + (size_t)B_ * L_ * E_;    // B*V (zeros)

  // Workspace carve-up
  char*  ws = (char*)d_ws;
  size_t off = 0;
  auto carve = [&](size_t bytes) -> void* {
    void* p = ws + off;
    off = (off + bytes + 255) & ~(size_t)255;
    return p;
  };
  float*  Wx      = (float*) carve((size_t)V_ * G4_ * 4);     // [V,4H]
  float*  gates   = (float*) carve((size_t)B_ * G4_ * 4);     // [B,4H]
  float*  logits  = (float*) carve((size_t)B_ * V_ * 4);      // [B,V]
  float*  hf      = (float*) carve((size_t)B_ * H_ * 4);
  float*  cst     = (float*) carve((size_t)B_ * H_ * 4);
  __bf16* hbf     = (__bf16*)carve((size_t)B_ * H_ * 2);
  __bf16* embf    = (__bf16*)carve((size_t)V_ * E_ * 2);
  __bf16* whhbf   = (__bf16*)carve((size_t)G4_ * H_ * 2);
  __bf16* wihbf   = (__bf16*)carve((size_t)G4_ * E_ * 2);
  __bf16* woutT   = (__bf16*)carve((size_t)V_ * H_ * 2);
  int*    idx     = (int*)   carve((size_t)B_ * 4);
  float*  ent_b   = (float*) carve((size_t)B_ * 4);

  // ---- init outputs ----
  zero_vec4_kernel<<<4096, 256, 0, stream>>>(out, N_OUT / 4);
  zero_scalar_kernel<<<4096, 256, 0, stream>>>(
      embeds_out, (size_t)B_ * L_ * E_ + (size_t)B_ * V_);
  (void)sent_prob;
  init_state_kernel<<<B_ / 256, 256, 0, stream>>>(out, seq_len, idx, ent_b);

  // ---- one-time precompute ----
  f32_to_bf16_kernel<<<1024, 256, 0, stream>>>(embedding, embf, (size_t)V_ * E_);
  f32_to_bf16_kernel<<<1024, 256, 0, stream>>>(W_hh, whhbf, (size_t)G4_ * H_);
  f32_to_bf16_kernel<<<1024, 256, 0, stream>>>(W_ih, wihbf, (size_t)G4_ * E_);
  transpose_to_bf16_kernel<<<(H_ * V_ + 255) / 256, 256, 0, stream>>>(W_out, woutT, H_, V_);
  h0_kernel<<<(B_ * H_) / 256, 256, 0, stream>>>(hidden_state, W_in, b_in, hf, hbf, cst);

  // Wx = embedding @ W_ih^T   (M=V, N=4H, K=E)
  gemm_bf16_wmma<<<dim3(G4_ / 128, V_ / 128), 256, 0, stream>>>(
      embf, wihbf, Wx, V_, G4_, E_);

  // ---- sequential decode ----
  for (int t = 0; t < L_; ++t) {
    // gates_h = h @ W_hh^T   (M=B, N=4H, K=H)
    gemm_bf16_wmma<<<dim3(G4_ / 128, B_ / 128), 256, 0, stream>>>(
        hbf, whhbf, gates, B_, G4_, H_);
    // embeds for this step use the incoming token index
    embeds_kernel<<<(B_ * E_) / 256, 256, 0, stream>>>(embedding, idx, embeds_out, t);
    // LSTM pointwise -> new h (f32 + bf16), c
    lstm_cell_kernel<<<(B_ * H_) / 256, 256, 0, stream>>>(
        gates, Wx, idx, b_ih, b_hh, cst, hf, hbf);
    // logits = h @ W_out   (M=B, N=V, K=H)
    gemm_bf16_wmma<<<dim3(V_ / 128, B_ / 128), 256, 0, stream>>>(
        hbf, woutT, logits, B_, V_, H_);
    // softmax stats + gumbel argmax + one-hot/seq_len/entropy
    softmax_argmax_kernel<<<B_, 256, 0, stream>>>(
        logits, b_out, gumbel + (size_t)t * B_ * V_, out, seq_len, idx, ent_b, t);
  }

  finalize_entropy_kernel<<<1, 256, 0, stream>>>(ent_b, ent_out);
}
